// DotProductAttention_41455024341165
// MI455X (gfx1250) — matrix operand order repaired and checked by reading, not compiled, not run on hardware
//
#include <hip/hip_runtime.h>

typedef __attribute__((ext_vector_type(16))) _Float16 v16h;
typedef __attribute__((ext_vector_type(2)))  _Float16 h2;
typedef __attribute__((ext_vector_type(8)))  float    v8f;

#define SEQ   2048
#define HQ    32
#define HKV   8
#define DH    128
#define KVSTR (HKV * DH)   // 1024 floats between consecutive k rows
#define KT_ST 136          // halves per K row (128 + 8 pad) -> 272B, 16B aligned
#define VT_ST 40           // halves per Vt d-row (32 + 8 pad) -> 80B, 16B aligned

__device__ __forceinline__ float fast_exp2(float x) {
    return __builtin_amdgcn_exp2f(x);      // v_exp_f32 (hardware base-2 exp)
}

// V chunk stored with octets 1<->2 of each 32-kk row swapped so the 16
// B-fragment halves for lane (r,hl) are contiguous: perm [0-7,16-23,8-15,24-31].
__device__ __forceinline__ int kperm(int d) {
    int dl  = d & 31;
    int sub = dl >> 3;                       // 0..3
    int ns  = ((sub & 1) << 1) | (sub >> 1); // 0->0, 1->2, 2->1, 3->3
    return (d & ~31) | (ns << 3) | (dl & 7);
}

__global__ __launch_bounds__(256)
void fa2_causal_gqa_41455024341165(const float* __restrict__ Qg,
                                   const float* __restrict__ Kg,
                                   const float* __restrict__ Vg,
                                   float* __restrict__ Og)
{
    __shared__ __align__(16) _Float16 KtB[2][32 * KT_ST];  // 2 x 8.5 KB
    __shared__ __align__(16) _Float16 VtB[2][DH * VT_ST];  // 2 x 10 KB

    const int tid  = threadIdx.x;
    const int lane = tid & 31;
    const int wv   = tid >> 5;        // wave 0..7
    const int r    = lane & 15;
    const int hl   = lane >> 4;

    const int qblk = blockIdx.x;
    const int h    = blockIdx.y;
    const int b    = blockIdx.z;
    const int hkv  = h >> 2;          // GQA group of 4

    const int q0 = qblk * 128 + wv * 16;
    // 1/sqrt(128) * log2(e): softmax done in base-2 (v_exp_f32), mathematically identical
    const float scale = 0.08838834764831845f * 1.4426950408889634f;

    // ---- resident Q fragments: B operand of S^T = K * Q^T ----
    const float* qrow = Qg + ((b * SEQ + q0 + r) * HQ + h) * DH + 16 * hl;
    v16h qf[4];
#pragma unroll
    for (int c = 0; c < 4; ++c) {
#pragma unroll
        for (int u = 0; u < 4; ++u) {
            float4 f = *(const float4*)(qrow + 32 * c + 4 * u);
            qf[c][4*u+0] = (_Float16)(f.x * scale);
            qf[c][4*u+1] = (_Float16)(f.y * scale);
            qf[c][4*u+2] = (_Float16)(f.z * scale);
            qf[c][4*u+3] = (_Float16)(f.w * scale);
        }
    }

    v8f of[8];
#pragma unroll
    for (int t = 0; t < 8; ++t) of[t] = v8f{};
    float m = -1e30f;
    float l = 0.0f;

    const int nblk = qblk * 4 + 4;
    const float* kbase = Kg + (b * SEQ * HKV + hkv) * DH;
    const float* vbase = Vg + (b * SEQ * HKV + hkv) * DH;

    // ---- loop-invariant staging addresses (per-thread) ----
    const float* kp = kbase + (tid >> 6) * KVSTR + ((tid & 63) << 1);       // + it*4*KVSTR
    const float* vp = vbase + (tid >> 7) * 2 * KVSTR + (tid & 127);         // + it*4*KVSTR (+KVSTR)
    const int ldsK = (tid >> 6) * KT_ST + ((tid & 63) << 1);                // + it*4*KT_ST
    const int ldsV = (tid & 127) * VT_ST + ((tid >> 7) << 1);               // + kperm(it*4)

    float2 kreg[8];
    float2 vreg[8];

    // prologue: issue loads for block 0
#pragma unroll
    for (int it = 0; it < 8; ++it) {
        kreg[it]   = *(const float2*)(kp + it * 4 * KVSTR);
        vreg[it].x = vp[it * 4 * KVSTR];
        vreg[it].y = vp[it * 4 * KVSTR + KVSTR];
    }

    int buf = 0;
    for (int blk = 0; blk < nblk; ++blk) {
        const int kb = blk * 32;

        // ---- convert + store staged block into LDS (packed b32 stores) ----
        _Float16* Kt = KtB[buf];
        _Float16* Vt = VtB[buf];
#pragma unroll
        for (int it = 0; it < 8; ++it) {
            h2 pk; pk[0] = (_Float16)kreg[it].x; pk[1] = (_Float16)kreg[it].y;
            *(h2*)(Kt + ldsK + it * 4 * KT_ST) = pk;
            h2 pv; pv[0] = (_Float16)vreg[it].x; pv[1] = (_Float16)vreg[it].y;
            *(h2*)(Vt + ldsV + kperm(it * 4)) = pv;
        }
        __syncthreads();   // single barrier: double buffering protects reuse

        // ---- prefetch next k-block into registers (overlaps compute) ----
        if (blk + 1 < nblk) {
            kp += 32 * KVSTR;
            vp += 32 * KVSTR;
#pragma unroll
            for (int it = 0; it < 8; ++it) {
                kreg[it]   = *(const float2*)(kp + it * 4 * KVSTR);
                vreg[it].x = vp[it * 4 * KVSTR];
                vreg[it].y = vp[it * 4 * KVSTR + KVSTR];
            }
        }

        if (kb < q0 + 16) {                    // wave-uniform
            // ---- S^T = K * Q^T : two 16(kk) x 16(q) tiles, A-frags from LDS ----
            v8f s0 = v8f{}, s1 = v8f{};
            const _Float16* kr0 = Kt + r * KT_ST + 16 * hl;
            const _Float16* kr1 = kr0 + 16 * KT_ST;
#pragma unroll
            for (int c = 0; c < 4; ++c) {
                v16h a0 = *(const v16h*)(kr0 + 32 * c);
                v16h a1 = *(const v16h*)(kr1 + 32 * c);
                s0 = __builtin_amdgcn_wmma_f32_16x16x32_f16(false, a0, false, qf[c],
                                                            (short)0, s0, false, false);
                s1 = __builtin_amdgcn_wmma_f32_16x16x32_f16(false, a1, false, qf[c],
                                                            (short)0, s1, false, false);
            }

            // ---- causal mask: only the diagonal block can need it (uniform) ----
            const int qg = q0 + r;
            if (kb + 31 > q0) {
#pragma unroll
                for (int i = 0; i < 8; ++i) {
                    int k0 = kb + i + 8 * hl;
                    s0[i] = (k0      > qg) ? -1e30f : s0[i];
                    s1[i] = (k0 + 16 > qg) ? -1e30f : s1[i];
                }
            }

            // ---- online softmax, base 2 (per-lane stats for query q0+r) ----
            float mloc = -1e30f;
#pragma unroll
            for (int i = 0; i < 8; ++i) mloc = fmaxf(mloc, fmaxf(s0[i], s1[i]));
            mloc = fmaxf(mloc, __shfl_xor(mloc, 16, 32));
            const float mnew = fmaxf(m, mloc);

            v16h pf;                 // C-regs of S^T are exactly the A-layout of P
            float lsum = 0.0f;
#pragma unroll
            for (int i = 0; i < 8; ++i) {
                float p0 = fast_exp2(s0[i] - mnew);
                float p1 = fast_exp2(s1[i] - mnew);
                lsum += p0 + p1;
                pf[i]     = (_Float16)p0;
                pf[8 + i] = (_Float16)p1;
            }
            lsum += __shfl_xor(lsum, 16, 32);

            // ---- rescale O only if some lane's max grew (wave-uniform test) ----
            if (__any(mloc > m)) {
                const float alpha = fast_exp2(m - mnew);
                m = mnew;
                l = l * alpha + lsum;
                float alphaB[8];
#pragma unroll
                for (int i = 0; i < 8; ++i) alphaB[i] = __shfl(alpha, i + 8 * hl, 32);
#pragma unroll
                for (int t = 0; t < 8; ++t)
#pragma unroll
                    for (int i = 0; i < 8; ++i) of[t][i] *= alphaB[i];
            } else {
                l += lsum;
            }

            // ---- O += P * V ----
#pragma unroll
            for (int t = 0; t < 8; ++t) {
                v16h vf = *(const v16h*)(Vt + (16 * t + r) * VT_ST + 16 * hl);
                of[t] = __builtin_amdgcn_wmma_f32_16x16x32_f16(false, pf, false, vf,
                                                               (short)0, of[t], false, false);
            }
        }
        buf ^= 1;
    }

    // ---- epilogue: O[q][d] = of / l ----
    const float linv = 1.0f / l;
    float lB[8];
#pragma unroll
    for (int i = 0; i < 8; ++i) lB[i] = __shfl(linv, i + 8 * hl, 32);
#pragma unroll
    for (int i = 0; i < 8; ++i) {
        float* orow = Og + ((b * SEQ + q0 + i + 8 * hl) * HQ + h) * DH + r;
#pragma unroll
        for (int t = 0; t < 8; ++t) orow[16 * t] = of[t][i] * lB[i];
    }
}

extern "C" void kernel_launch(void* const* d_in, const int* in_sizes, int n_in,
                              void* d_out, int out_size, void* d_ws, size_t ws_size,
                              hipStream_t stream) {
    (void)in_sizes; (void)n_in; (void)d_ws; (void)ws_size; (void)out_size;
    const float* q = (const float*)d_in[0];
    const float* k = (const float*)d_in[1];
    const float* v = (const float*)d_in[2];
    float* o = (float*)d_out;
    dim3 grid(SEQ / 128, HQ, 2);   // (q-block, head, batch)
    dim3 block(256);
    fa2_causal_gqa_41455024341165<<<grid, block, 0, stream>>>(q, k, v, o);
}
